// BaseSelfAttention_88433376625006
// MI455X (gfx1250) — matrix operand order repaired and checked by reading, not compiled, not run on hardware
//
#include <hip/hip_runtime.h>

typedef __attribute__((ext_vector_type(16))) _Float16 v16h;
typedef __attribute__((ext_vector_type(8)))  float    v8f;

union HFrag { v16h v; unsigned u[8]; _Float16 h[16]; };

#define NN   4096
#define DD   768
#define HEADS 12
#define DH   64

__device__ inline v8f wmma32(v16h a, v16h b, v8f c) {
  return __builtin_amdgcn_wmma_f32_16x16x32_f16(false, a, false, b, (short)0, c, false, false);
}

// A fragment: 16x32 f16 tile from row-major src (ld = elements per row).
// ISA layout: lane m=lane&15, half=lane>>4; VGPR i holds K = 16*(i>>2) + 8*half + 2*(i&3), +1
__device__ inline v16h load_a_frag(const _Float16* src, int ld, int row0, int k0) {
  int lane = threadIdx.x & 31;
  int m = lane & 15, hf = lane >> 4;
  const _Float16* rp = src + (size_t)(row0 + m) * ld + k0;
  HFrag f;
#pragma unroll
  for (int i = 0; i < 8; ++i) {
    int k = ((i >> 2) << 4) + (hf << 3) + ((i & 3) << 1);
    f.u[i] = *(const unsigned*)(rp + k);
  }
  return f.v;
}

// B fragment: 32x16 where memory holds B^T, i.e. bt[n][k] row-major with stride ld.
// ISA layout: lane col n=lane&15, K base = 8*(lane>>4); VGPR i holds K = base + 16*(i>>2) + 2*(i&3), +1
__device__ inline v16h load_b_frag_t(const _Float16* bt, int ld, int n0, int k0) {
  int lane = threadIdx.x & 31;
  int n = n0 + (lane & 15);
  int ks = k0 + ((lane >> 4) << 3);
  const _Float16* rp = bt + (size_t)n * ld + ks;
  HFrag f;
#pragma unroll
  for (int i = 0; i < 8; ++i) f.u[i] = *(const unsigned*)(rp + ((i >> 2) << 4) + ((i & 3) << 1));
  return f.v;
}

// ---------------- LayerNorm: x[4096,768] f32 -> xn f16 ----------------
__global__ void ln_kernel(const float* __restrict__ x, const float* __restrict__ g,
                          const float* __restrict__ b, _Float16* __restrict__ xn) {
  int row = blockIdx.x;
  const float* xr = x + (size_t)row * DD;
  __shared__ float red[256];
  float s = 0.f;
  for (int i = threadIdx.x; i < DD; i += 256) s += xr[i];
  red[threadIdx.x] = s; __syncthreads();
  for (int off = 128; off > 0; off >>= 1) {
    if (threadIdx.x < off) red[threadIdx.x] += red[threadIdx.x + off];
    __syncthreads();
  }
  float mu = red[0] * (1.0f / DD);
  __syncthreads();
  float v = 0.f;
  for (int i = threadIdx.x; i < DD; i += 256) { float d = xr[i] - mu; v += d * d; }
  red[threadIdx.x] = v; __syncthreads();
  for (int off = 128; off > 0; off >>= 1) {
    if (threadIdx.x < off) red[threadIdx.x] += red[threadIdx.x + off];
    __syncthreads();
  }
  float rstd = rsqrtf(red[0] * (1.0f / DD) + 1e-5f);
  for (int i = threadIdx.x; i < DD; i += 256)
    xn[(size_t)row * DD + i] = (_Float16)((xr[i] - mu) * rstd * g[i] + b[i]);
}

// ---------------- weight prep: f32 [K,N] -> f16 transposed [N,K] ----------------
__global__ void prep_w_kernel(const float* __restrict__ w, _Float16* __restrict__ wt,
                              int K, int N) {
  int idx = blockIdx.x * 256 + threadIdx.x;
  if (idx >= K * N) return;
  int k = idx / N, n = idx % N;
  wt[(size_t)n * K + k] = (_Float16)w[idx];
}

// ---------------- QKV GEMM: XN[4096,768] @ Wt -> Q/K [12][4096][64], Vt [12][64][4096] ----------------
__global__ void qkv_gemm_kernel(const _Float16* __restrict__ xn, const _Float16* __restrict__ wt,
                                _Float16* __restrict__ Q, _Float16* __restrict__ K,
                                _Float16* __restrict__ Vt) {
  int row0 = blockIdx.x * 32;
  int col0 = blockIdx.y * 64;
  v8f acc[2][4] = {};
  for (int kb = 0; kb < DD; kb += 32) {
    v16h a0 = load_a_frag(xn, DD, row0,      kb);
    v16h a1 = load_a_frag(xn, DD, row0 + 16, kb);
#pragma unroll
    for (int t = 0; t < 4; ++t) {
      v16h bf = load_b_frag_t(wt, DD, col0 + t * 16, kb);
      acc[0][t] = wmma32(a0, bf, acc[0][t]);
      acc[1][t] = wmma32(a1, bf, acc[1][t]);
    }
  }
  int lane = threadIdx.x & 31;
  int cl = lane & 15, hf = lane >> 4;
#pragma unroll
  for (int t = 0; t < 4; ++t) {
    int col = col0 + t * 16 + cl;
    int which = col / DD;          // 0=q 1=k 2=v
    int h = (col % DD) / DH;
    int d = col % DH;
#pragma unroll
    for (int qt = 0; qt < 2; ++qt) {
#pragma unroll
      for (int r = 0; r < 8; ++r) {
        int row = row0 + qt * 16 + r + 8 * hf;
        float val = acc[qt][t][r];
        if (which == 0) {
          Q[((size_t)h * NN + row) * DH + d] = (_Float16)(val * 0.125f);  // Dh^-0.5
        } else if (which == 1) {
          K[((size_t)h * NN + row) * DH + d] = (_Float16)val;
        } else {
          Vt[((size_t)h * DH + d) * NN + row] = (_Float16)val;            // transposed
        }
      }
    }
  }
}

// ---------------- causal flash attention: one wave per (head, 32-query block) ----------------
__global__ void attn_kernel(const _Float16* __restrict__ Q, const _Float16* __restrict__ K,
                            const _Float16* __restrict__ Vt, _Float16* __restrict__ att) {
  int h = blockIdx.y;
  int qbase = blockIdx.x * 32;
  const _Float16* Qh  = Q  + (size_t)h * NN * DH;
  const _Float16* Kh  = K  + (size_t)h * NN * DH;
  const _Float16* Vth = Vt + (size_t)h * DH * NN;
  __shared__ _Float16 pt[32 * 32];

  int lane = threadIdx.x & 31;
  int cl = lane & 15, hf = lane >> 4;

  v16h qa[2][2];
  qa[0][0] = load_a_frag(Qh, DH, qbase,      0);
  qa[0][1] = load_a_frag(Qh, DH, qbase,      32);
  qa[1][0] = load_a_frag(Qh, DH, qbase + 16, 0);
  qa[1][1] = load_a_frag(Qh, DH, qbase + 16, 32);

  v8f o[2][4] = {};
  float m[2][8], ssum[2][8];
#pragma unroll
  for (int qt = 0; qt < 2; ++qt)
#pragma unroll
    for (int r = 0; r < 8; ++r) { m[qt][r] = -3e38f; ssum[qt][r] = 0.f; }

  int jend = qbase + 32;
  for (int jb = 0; jb < jend; jb += 32) {
    // prefetch next key block (speculative; per-lane covers the 32 rows / d-lines)
    __builtin_prefetch(Kh + (size_t)(jb + 32 + lane) * DH, 0, 0);
    __builtin_prefetch(Vth + (size_t)lane * NN + jb + 32, 0, 0);

    // S = Qs * K^T  (K row-major [j][d] acts as B^T)
    v16h k0a = load_b_frag_t(Kh, DH, jb,      0);
    v16h k0b = load_b_frag_t(Kh, DH, jb,      32);
    v16h k1a = load_b_frag_t(Kh, DH, jb + 16, 0);
    v16h k1b = load_b_frag_t(Kh, DH, jb + 16, 32);

#pragma unroll
    for (int qt = 0; qt < 2; ++qt) {
      v8f s0 = {}; s0 = wmma32(qa[qt][0], k0a, s0); s0 = wmma32(qa[qt][1], k0b, s0);
      v8f s1 = {}; s1 = wmma32(qa[qt][0], k1a, s1); s1 = wmma32(qa[qt][1], k1b, s1);
      int j0 = jb + cl, j1 = j0 + 16;
#pragma unroll
      for (int r = 0; r < 8; ++r) {
        int row = qbase + qt * 16 + r + 8 * hf;
        float a0 = (j0 > row) ? -3e38f : s0[r];
        float a1 = (j1 > row) ? -3e38f : s1[r];
        // row max across the 16 lanes of this half
        float t = fmaxf(a0, a1);
        t = fmaxf(t, __shfl_xor(t, 1, 16));
        t = fmaxf(t, __shfl_xor(t, 2, 16));
        t = fmaxf(t, __shfl_xor(t, 4, 16));
        t = fmaxf(t, __shfl_xor(t, 8, 16));
        float mn = fmaxf(m[qt][r], t);
        float alpha = __expf(m[qt][r] - mn);
        m[qt][r] = mn;
        float p0 = __expf(a0 - mn);
        float p1 = __expf(a1 - mn);
        float rs = p0 + p1;
        rs += __shfl_xor(rs, 1, 16);
        rs += __shfl_xor(rs, 2, 16);
        rs += __shfl_xor(rs, 4, 16);
        rs += __shfl_xor(rs, 8, 16);
        ssum[qt][r] = ssum[qt][r] * alpha + rs;
        o[qt][0][r] *= alpha; o[qt][1][r] *= alpha;
        o[qt][2][r] *= alpha; o[qt][3][r] *= alpha;
        int rr = qt * 16 + r + 8 * hf;
        pt[rr * 32 + cl]      = (_Float16)p0;
        pt[rr * 32 + cl + 16] = (_Float16)p1;
      }
    }
    __syncthreads();
    v16h pa0 = load_a_frag(pt, 32, 0,  0);
    v16h pa1 = load_a_frag(pt, 32, 16, 0);
    __syncthreads();
    // O += P * V using transposed V: B^T[n=d][k=key] = Vt[d][j], contiguous dwords
#pragma unroll
    for (int t = 0; t < 4; ++t) {
      v16h vb = load_b_frag_t(Vth, NN, t * 16, jb);
      o[0][t] = wmma32(pa0, vb, o[0][t]);
      o[1][t] = wmma32(pa1, vb, o[1][t]);
    }
  }

  // normalize and store into att[n, h*64 + d] (fp16)
#pragma unroll
  for (int qt = 0; qt < 2; ++qt)
#pragma unroll
    for (int r = 0; r < 8; ++r) {
      float inv = 1.0f / ssum[qt][r];
      int row = qbase + qt * 16 + r + 8 * hf;
      size_t base = (size_t)row * DD + h * DH;
      att[base +  0 + cl] = (_Float16)(o[qt][0][r] * inv);
      att[base + 16 + cl] = (_Float16)(o[qt][1][r] * inv);
      att[base + 32 + cl] = (_Float16)(o[qt][2][r] * inv);
      att[base + 48 + cl] = (_Float16)(o[qt][3][r] * inv);
    }
}

// ---------------- output projection: ATT[4096,768] @ WoutT -> out f32 ----------------
__global__ void out_gemm_kernel(const _Float16* __restrict__ att, const _Float16* __restrict__ wt,
                                float* __restrict__ out) {
  int row0 = blockIdx.x * 32;
  int col0 = blockIdx.y * 64;
  v8f acc[2][4] = {};
  for (int kb = 0; kb < DD; kb += 32) {
    v16h a0 = load_a_frag(att, DD, row0,      kb);
    v16h a1 = load_a_frag(att, DD, row0 + 16, kb);
#pragma unroll
    for (int t = 0; t < 4; ++t) {
      v16h bf = load_b_frag_t(wt, DD, col0 + t * 16, kb);
      acc[0][t] = wmma32(a0, bf, acc[0][t]);
      acc[1][t] = wmma32(a1, bf, acc[1][t]);
    }
  }
  int lane = threadIdx.x & 31;
  int cl = lane & 15, hf = lane >> 4;
#pragma unroll
  for (int qt = 0; qt < 2; ++qt)
#pragma unroll
    for (int t = 0; t < 4; ++t)
#pragma unroll
      for (int r = 0; r < 8; ++r) {
        int row = row0 + qt * 16 + r + 8 * hf;
        out[(size_t)row * DD + col0 + t * 16 + cl] = acc[qt][t][r];
      }
}

extern "C" void kernel_launch(void* const* d_in, const int* in_sizes, int n_in,
                              void* d_out, int out_size, void* d_ws, size_t ws_size,
                              hipStream_t stream) {
  const float* x     = (const float*)d_in[0];
  const float* ln_g  = (const float*)d_in[1];
  const float* ln_b  = (const float*)d_in[2];
  const float* w_qkv = (const float*)d_in[3];
  const float* w_out = (const float*)d_in[4];
  // d_in[5] = mask, all-true -> ignored (causal handled in-kernel)

  char* ws = (char*)d_ws;
  _Float16* XN    = (_Float16*)(ws);                  //  6,291,456 B
  _Float16* WQKVt = (_Float16*)(ws +  6291456);       //  3,538,944 B
  _Float16* WOUTt = (_Float16*)(ws +  9830400);       //  1,179,648 B
  _Float16* Qb    = (_Float16*)(ws + 11010048);       //  6,291,456 B
  _Float16* Kb    = (_Float16*)(ws + 17301504);       //  6,291,456 B
  _Float16* Vtb   = (_Float16*)(ws + 23592960);       //  6,291,456 B (transposed V)
  _Float16* ATT   = (_Float16*)(ws + 29884416);       //  6,291,456 B -> 36,175,872 total
  float* out = (float*)d_out;

  ln_kernel<<<NN, 256, 0, stream>>>(x, ln_g, ln_b, XN);
  prep_w_kernel<<<(DD * 3 * DD + 255) / 256, 256, 0, stream>>>(w_qkv, WQKVt, DD, 3 * DD);
  prep_w_kernel<<<(DD * DD + 255) / 256, 256, 0, stream>>>(w_out, WOUTt, DD, DD);
  qkv_gemm_kernel<<<dim3(NN / 32, (3 * DD) / 64), 32, 0, stream>>>(XN, WQKVt, Qb, Kb, Vtb);
  attn_kernel<<<dim3(NN / 32, HEADS), 32, 0, stream>>>(Qb, Kb, Vtb, ATT);
  out_gemm_kernel<<<dim3(NN / 32, DD / 64), 32, 0, stream>>>(ATT, WOUTt, out);
}
